// ContractiveInvertibleGNN_78099685310586
// MI455X (gfx1250) — matrix operand order repaired and checked by reading, not compiled, not run on hardware
//
#include <hip/hip_runtime.h>
#include <hip/hip_bf16.h>

// ---------------------------------------------------------------------------
// ContractiveInvertibleGNN forward, fused for MI455X (gfx1250, wave32, WMMA).
//
//   g layer1:  h1[b,i,:] = lrelu(X[b,i]*g_W1[i,:] + c_g[i,:])      (elementwise)
//   out[b,i]  = h2f[b,i,:] . f_W3[:,i] + f_b3[i]                   (dot product)
// All interior GEMMs on v_wmma_f32_16x16x32_f16 (f32 accumulate).
//
// One wave = one batch element (its 32 node rows); the node aggregation never
// leaves the wave.  Shared weights are staged in LDS once per workgroup
// (single barrier), so the hot GEMM loops read operands exclusively from LDS.
// The 32x32 node-feature tile (X_emb / X_aggr) lives in spare columns 96..127
// of the 32x136 activation buffer (g's h2 is dead by the time it is written).
// ---------------------------------------------------------------------------

typedef __attribute__((ext_vector_type(16))) _Float16 v16h;
typedef __attribute__((ext_vector_type(8)))  _Float16 v8h;
typedef __attribute__((ext_vector_type(8)))  float    v8f;

union F16x16 { v16h v; v8h h[2]; };

#define WMMA_F16(a, b, c) \
  __builtin_amdgcn_wmma_f32_16x16x32_f16(false, (a), false, (b), (short)0, (c), false, false)

// A-fragment (16x32 f16): lane<16 -> row lane, k = {kB..kB+7, kB+16..kB+23};
// lane>=16 -> row lane-16, k = {kB+8..kB+15, kB+24..kB+31}.
__device__ __forceinline__ v16h load_afrag(const _Float16* base, int stride,
                                           int rowBase, int kBase, int lane) {
  int hi = lane >> 4, lr = lane & 15;
  const _Float16* p = base + (rowBase + lr) * stride + kBase + hi * 8;
  F16x16 f;
  f.h[0] = *(const v8h*)p;
  f.h[1] = *(const v8h*)(p + 16);
  return f.v;
}

// B-fragment (32x16 f16): lane -> column nB+lane%16; k = {kB+8h..kB+8h+7, kB+16+8h..kB+23+8h}.
__device__ __forceinline__ v16h load_bfrag(const _Float16* base, int stride,
                                           int kBase, int nBase, int lane) {
  int hi = lane >> 4, lr = lane & 15;
  const _Float16* p = base + (kBase + 8 * hi) * stride + nBase + lr;
  v16h f;
#pragma unroll
  for (int s = 0; s < 8; ++s) {
    f[s] = p[s * stride];
    f[s + 8] = p[(16 + s) * stride];
  }
  return f;
}

__device__ __forceinline__ float lrelu(float t) {
  return fmaf(0.01f, fminf(t, 0.f), fmaxf(t, 0.f));
}

// H = H + lrelu(H @ W(128x128) + bias); W staged in LDS.  Both 16-row M-tiles
// are accumulated together: B-fragments are loaded once and the two WMMA
// chains are independent (better XDL pipelining).
__device__ __forceinline__ void residual_block(_Float16 (*A)[136],
                                               const _Float16 (*Ws)[136],
                                               const float* __restrict__ bias,
                                               int lane) {
  int hi = lane >> 4, lr = lane & 15;
  v16h a[2][4];
#pragma unroll
  for (int mt = 0; mt < 2; ++mt)
#pragma unroll
    for (int kc = 0; kc < 4; ++kc)
      a[mt][kc] = load_afrag(&A[0][0], 136, mt * 16, kc * 32, lane);
  for (int nt = 0; nt < 8; ++nt) {
    v8f acc0 = {}, acc1 = {};
#pragma unroll
    for (int kc = 0; kc < 4; ++kc) {
      v16h bf = load_bfrag(&Ws[0][0], 136, kc * 32, nt * 16, lane);
      acc0 = WMMA_F16(a[0][kc], bf, acc0);
      acc1 = WMMA_F16(a[1][kc], bf, acc1);
    }
    int col = nt * 16 + lr;
    float bb = bias[col];
#pragma unroll
    for (int r = 0; r < 8; ++r) {
      int row0 = r + 8 * hi, row1 = 16 + r + 8 * hi;
      float t0 = lrelu(acc0[r] + bb);
      float t1 = lrelu(acc1[r] + bb);
      A[row0][col] = (_Float16)((float)A[row0][col] + t0);
      A[row1][col] = (_Float16)((float)A[row1][col] + t1);
    }
  }
}

// ---------------------------------------------------------------------------
// Prep kernel: per-node constants + f16 weight copies into workspace.
// ---------------------------------------------------------------------------
__global__ void gnn_prep(const float* __restrict__ W, const float* __restrict__ embd,
                         const float* __restrict__ gW1, const float* __restrict__ gb1,
                         const float* __restrict__ gW2, const float* __restrict__ gW3,
                         const float* __restrict__ fW1, const float* __restrict__ fb1,
                         const float* __restrict__ fW2,
                         float* __restrict__ cg, float* __restrict__ cf,
                         _Float16* __restrict__ gW2h, _Float16* __restrict__ fW2h,
                         _Float16* __restrict__ gW3h, _Float16* __restrict__ fW1ah,
                         _Float16* __restrict__ WTh) {
  int tid = blockIdx.x * blockDim.x + threadIdx.x;
  int nthr = gridDim.x * blockDim.x;
  // c_g[i,k] = emb[i,:] @ g_W1[32:64, k] + g_b1[k]; c_f likewise with f_W1/f_b1.
  for (int idx = tid; idx < 32 * 128; idx += nthr) {
    int i = idx >> 7, k = idx & 127;
    float sg = gb1[k], sf = fb1[k];
    for (int e = 0; e < 32; ++e) {
      float em = embd[i * 32 + e];
      sg = fmaf(em, gW1[(32 + e) * 128 + k], sg);
      sf = fmaf(em, fW1[(32 + e) * 128 + k], sf);
    }
    cg[idx] = sg;
    cf[idx] = sf;
  }
  for (int idx = tid; idx < 128 * 128; idx += nthr) {
    gW2h[idx] = (_Float16)gW2[idx];
    fW2h[idx] = (_Float16)fW2[idx];
  }
  for (int idx = tid; idx < 128 * 32; idx += nthr) {
    gW3h[idx] = (_Float16)gW3[idx];
    fW1ah[idx] = (_Float16)fW1[idx];  // first 32 rows of f_W1
  }
  for (int idx = tid; idx < 32 * 32; idx += nthr) {
    int i = idx >> 5, j = idx & 31;
    WTh[idx] = (_Float16)((i == j) ? 0.f : W[j * 32 + i]);  // W_adj^T, diag zeroed
  }
}

// ---------------------------------------------------------------------------
// Main fused kernel: 8 waves/WG, one batch element per wave.
// ---------------------------------------------------------------------------
__global__ __launch_bounds__(256) void gnn_main(
    const float* __restrict__ X, const float* __restrict__ gW1,
    const float* __restrict__ gb2, const float* __restrict__ gb3,
    const float* __restrict__ fb2, const float* __restrict__ fW3,
    const float* __restrict__ fb3,
    const float* __restrict__ cg, const float* __restrict__ cf,
    const _Float16* __restrict__ gW2h, const _Float16* __restrict__ fW2h,
    const _Float16* __restrict__ gW3h, const _Float16* __restrict__ fW1ah,
    const _Float16* __restrict__ WTh,
    float* __restrict__ out, int Bn) {
  __shared__ __align__(16) _Float16 act_s[8][32][136];  // per-wave activations
  __shared__ __align__(16) _Float16 sGW2[128][136];     // g_W2 (f16, padded)
  __shared__ __align__(16) _Float16 sFW2[128][136];     // f_W2
  __shared__ __align__(16) _Float16 sGW3[128][40];      // g_W3
  __shared__ __align__(16) _Float16 sFW1[32][136];      // f_W1[0:32,:]

  int tid = threadIdx.x;
  // ---- cooperative weight staging (global f16 -> LDS), one barrier ----
  for (int idx = tid; idx < 2048; idx += 256) {  // 128x128 in v8h chunks
    int r = idx >> 4, c = (idx & 15) * 8;
    *(v8h*)&sGW2[r][c] = *(const v8h*)&gW2h[r * 128 + c];
    *(v8h*)&sFW2[r][c] = *(const v8h*)&fW2h[r * 128 + c];
  }
  for (int idx = tid; idx < 512; idx += 256) {   // 128x32
    int r = idx >> 2, c = (idx & 3) * 8;
    *(v8h*)&sGW3[r][c] = *(const v8h*)&gW3h[r * 32 + c];
  }
  for (int idx = tid; idx < 512; idx += 256) {   // 32x128
    int r = idx >> 4, c = (idx & 15) * 8;
    *(v8h*)&sFW1[r][c] = *(const v8h*)&fW1ah[r * 128 + c];
  }
  __syncthreads();

  int wave = tid >> 5;
  int lane = tid & 31;
  int b = blockIdx.x * 8 + wave;
  if (b >= Bn) return;  // wave-uniform (after the only barrier)

  _Float16 (*A)[136] = act_s[wave];
  _Float16* Eb = &A[0][96];  // 32x32 node tile view, stride 136
  int hi = lane >> 4, lr = lane & 15;

  // ---- g layer 1 (elementwise): h1[i,k] = lrelu(X[b,i]*gW1[i,k] + cg[i,k])
  for (int i = 0; i < 32; ++i) {
    float xi = X[b * 32 + i];
#pragma unroll
    for (int c = 0; c < 2; ++c) {
      int k = c * 64 + lane * 2;
      A[i][k]     = (_Float16)lrelu(fmaf(xi, gW1[i * 128 + k],     cg[i * 128 + k]));
      A[i][k + 1] = (_Float16)lrelu(fmaf(xi, gW1[i * 128 + k + 1], cg[i * 128 + k + 1]));
    }
  }

  // ---- g layer 2: residual 128x128 block (WMMA, weights in LDS)
  residual_block(A, sGW2, gb2, lane);

  // ---- g layer 3: X_emb = H @ g_W3 (128->32) + g_b3  -> Eb (cols 96..127)
  {
    v16h a[2][4];
#pragma unroll
    for (int mt = 0; mt < 2; ++mt)
#pragma unroll
      for (int kc = 0; kc < 4; ++kc)
        a[mt][kc] = load_afrag(&A[0][0], 136, mt * 16, kc * 32, lane);
#pragma unroll
    for (int nt = 0; nt < 2; ++nt) {
      v8f acc0 = {}, acc1 = {};
#pragma unroll
      for (int kc = 0; kc < 4; ++kc) {
        v16h bf = load_bfrag(&sGW3[0][0], 40, kc * 32, nt * 16, lane);
        acc0 = WMMA_F16(a[0][kc], bf, acc0);
        acc1 = WMMA_F16(a[1][kc], bf, acc1);
      }
      int col = nt * 16 + lr;
      float bb = gb3[col];
#pragma unroll
      for (int r = 0; r < 8; ++r) {
        Eb[(r + 8 * hi) * 136 + col]      = (_Float16)(acc0[r] + bb);
        Eb[(16 + r + 8 * hi) * 136 + col] = (_Float16)(acc1[r] + bb);
      }
    }
  }

  // ---- aggregation: X_aggr = W_adj^T (32x32) @ X_emb (32x32), K=32 WMMA
  v8f agg[2][2];
  for (int mt = 0; mt < 2; ++mt) {
    v16h wfr = load_afrag(WTh, 32, mt * 16, 0, lane);
#pragma unroll
    for (int nt = 0; nt < 2; ++nt) {
      v16h bf = load_bfrag(Eb, 136, 0, nt * 16, lane);
      v8f z = {};
      agg[mt][nt] = WMMA_F16(wfr, bf, z);
    }
  }
  // overwrite Eb with X_aggr (all Eb reads above are done; same-wave LDS in order)
#pragma unroll
  for (int mt = 0; mt < 2; ++mt)
#pragma unroll
    for (int nt = 0; nt < 2; ++nt) {
      int col = nt * 16 + lr;
#pragma unroll
      for (int r = 0; r < 8; ++r)
        Eb[(mt * 16 + r + 8 * hi) * 136 + col] = (_Float16)agg[mt][nt][r];
    }

  // ---- f layer 1: H = lrelu(X_aggr @ f_W1[0:32,:] + c_f)  (32->128)
  {
    v16h a0 = load_afrag(Eb, 136, 0, 0, lane);    // A-frags cached before any
    v16h a1 = load_afrag(Eb, 136, 16, 0, lane);   // writes clobber Eb cols
    for (int nt = 0; nt < 8; ++nt) {
      v16h bf = load_bfrag(&sFW1[0][0], 136, 0, nt * 16, lane);
      v8f z0 = {}, z1 = {};
      v8f acc0 = WMMA_F16(a0, bf, z0);
      v8f acc1 = WMMA_F16(a1, bf, z1);
      int col = nt * 16 + lr;
#pragma unroll
      for (int r = 0; r < 8; ++r) {
        int row0 = r + 8 * hi, row1 = 16 + r + 8 * hi;
        A[row0][col] = (_Float16)lrelu(acc0[r] + cf[row0 * 128 + col]);
        A[row1][col] = (_Float16)lrelu(acc1[r] + cf[row1 * 128 + col]);
      }
    }
  }

  // ---- f layer 2: residual 128x128 block (WMMA, weights in LDS)
  residual_block(A, sFW2, fb2, lane);

  // ---- f layer 3: out[b,i] = H[i,:] . f_W3[:,i] + f_b3[i]  (lane i dot)
  float accd = 0.f;
#pragma unroll 4
  for (int kc = 0; kc < 16; ++kc) {
    v8h chunk = *(const v8h*)&A[lane][kc * 8];
#pragma unroll
    for (int j = 0; j < 8; ++j)
      accd = fmaf((float)chunk[j], fW3[(kc * 8 + j) * 32 + lane], accd);
  }
  out[b * 32 + lane] = accd + fb3[lane];
}

// ---------------------------------------------------------------------------
extern "C" void kernel_launch(void* const* d_in, const int* in_sizes, int n_in,
                              void* d_out, int out_size, void* d_ws, size_t ws_size,
                              hipStream_t stream) {
  const float* X    = (const float*)d_in[0];
  const float* W    = (const float*)d_in[1];
  const float* embd = (const float*)d_in[2];
  const float* gW1  = (const float*)d_in[3];
  const float* gb1  = (const float*)d_in[4];
  const float* gW2  = (const float*)d_in[5];
  const float* gb2  = (const float*)d_in[6];
  const float* gW3  = (const float*)d_in[7];
  const float* gb3  = (const float*)d_in[8];
  const float* fW1  = (const float*)d_in[9];
  const float* fb1  = (const float*)d_in[10];
  const float* fW2  = (const float*)d_in[11];
  const float* fb2  = (const float*)d_in[12];
  const float* fW3  = (const float*)d_in[13];
  const float* fb3  = (const float*)d_in[14];
  float* out = (float*)d_out;

  // workspace layout (114 KB total)
  char* ws = (char*)d_ws;
  float*    cg    = (float*)(ws + 0);
  float*    cf    = (float*)(ws + 16 * 1024);
  _Float16* gW2h  = (_Float16*)(ws + 32 * 1024);
  _Float16* fW2h  = (_Float16*)(ws + 64 * 1024);
  _Float16* gW3h  = (_Float16*)(ws + 96 * 1024);
  _Float16* fW1ah = (_Float16*)(ws + 104 * 1024);
  _Float16* WTh   = (_Float16*)(ws + 112 * 1024);

  int Bn = in_sizes[0] / 32;

  gnn_prep<<<64, 256, 0, stream>>>(W, embd, gW1, gb1, gW2, gW3, fW1, fb1, fW2,
                                   cg, cf, gW2h, fW2h, gW3h, fW1ah, WTh);
  gnn_main<<<(Bn + 7) / 8, 256, 0, stream>>>(X, gW1, gb2, gb3, fb2, fW3, fb3,
                                             cg, cf, gW2h, fW2h, gW3h, fW1ah,
                                             WTh, out, Bn);
}
